// HubertEncoder_87823491268904
// MI455X (gfx1250) — matrix operand order repaired and checked by reading, not compiled, not run on hardware
//
#include <hip/hip_runtime.h>

typedef __attribute__((ext_vector_type(16))) __bf16 v16bf;
typedef __attribute__((ext_vector_type(8)))  __bf16 v8bf;
typedef __attribute__((ext_vector_type(8)))  float  v8f;
typedef __attribute__((ext_vector_type(4)))  int    v4i;

#define D_DIM   768
#define K_STEPS 24      // 768 / 32
#define K_CODES 2048
#define N_ROWS  48000   // B*T = 32*1500

// ---- LDS layout for the GEMM kernel (dynamic shared memory) ----
#define ROW_BYTES       1536                  // 768 bf16 per code row
#define LDS_ROW_STRIDE  1552                  // +16B pad: 388 dwords % 64 banks = 4 -> conflict-free
#define LDS_PLANE       (16 * LDS_ROW_STRIDE) // 24832 B (one 16-code strip, one plane)
#define LDS_BUF         (2 * LDS_PLANE)       // 49664 B (hi+lo planes)
#define LDS_CSQ         8192                  // 2048 f32
#define SMEM_BYTES      (LDS_CSQ + 2 * LDS_BUF) // 107520 B  (<< 320KB WGP LDS)

// ---- CDNA5 async global->LDS path (guarded; falls back to load+ds_store) ----
#if defined(__HIP_DEVICE_COMPILE__) && \
    __has_builtin(__builtin_amdgcn_global_load_async_to_lds_b128) && \
    __has_builtin(__builtin_amdgcn_s_wait_asynccnt)
#define HAS_ASYNC_LDS 1
#else
#define HAS_ASYNC_LDS 0
#endif

__device__ __forceinline__ void async_cp16(const char* g, char* l) {
#if HAS_ASYNC_LDS
    __builtin_amdgcn_global_load_async_to_lds_b128(
        (__attribute__((address_space(1))) v4i*)g,
        (__attribute__((address_space(3))) v4i*)l, 0, 0);
#else
    *(v8bf*)l = *(const v8bf*)g;
#endif
}

#if HAS_ASYNC_LDS
#define WAIT_ASYNC(n) __builtin_amdgcn_s_wait_asynccnt(n)
#else
#define WAIT_ASYNC(n) do { } while (0)
#endif

// ---------------- Kernel 1: LayerNorm rows -> bf16 hi/lo planes ----------------
__global__ __launch_bounds__(256) void ln_split_kernel(
    const float* __restrict__ E, __bf16* __restrict__ ehi, __bf16* __restrict__ elo)
{
    const int row = blockIdx.x;
    const float* x = E + (size_t)row * D_DIM;
    const int tid = threadIdx.x;

    float v0 = x[tid], v1 = x[tid + 256], v2 = x[tid + 512];
    float s = v0 + v1 + v2;
    float q = v0 * v0 + v1 * v1 + v2 * v2;

    #pragma unroll
    for (int off = 16; off > 0; off >>= 1) {
        s += __shfl_xor(s, off);
        q += __shfl_xor(q, off);
    }
    __shared__ float ss[8], sq[8];
    const int w = tid >> 5, ln = tid & 31;
    if (ln == 0) { ss[w] = s; sq[w] = q; }
    __syncthreads();
    if (w == 0) {
        float a = (ln < 8) ? ss[ln] : 0.f;
        float b = (ln < 8) ? sq[ln] : 0.f;
        #pragma unroll
        for (int off = 4; off > 0; off >>= 1) {
            a += __shfl_xor(a, off);
            b += __shfl_xor(b, off);
        }
        if (ln == 0) { ss[0] = a; sq[0] = b; }
    }
    __syncthreads();

    const float mu  = ss[0] * (1.0f / D_DIM);
    const float var = sq[0] * (1.0f / D_DIM) - mu * mu;
    const float rs  = rsqrtf(var + 1e-5f);

    __bf16* oh = ehi + (size_t)row * D_DIM;
    __bf16* ol = elo + (size_t)row * D_DIM;
    float vals[3] = { v0, v1, v2 };
    #pragma unroll
    for (int i = 0; i < 3; ++i) {
        const int d = tid + i * 256;
        float v  = (vals[i] - mu) * rs;
        __bf16 h = (__bf16)v;
        oh[d] = h;
        ol[d] = (__bf16)(v - (float)h);
    }
}

// -------- Kernel 2: split C -> bf16 hi/lo, compute ||C_k||^2 (f32, exact) --------
__global__ __launch_bounds__(256) void prep_c_kernel(
    const float* __restrict__ C, __bf16* __restrict__ chi, __bf16* __restrict__ clo,
    float* __restrict__ csq)
{
    const int row = blockIdx.x;
    const float* x = C + (size_t)row * D_DIM;
    const int tid = threadIdx.x;

    float v0 = x[tid], v1 = x[tid + 256], v2 = x[tid + 512];
    float q = v0 * v0 + v1 * v1 + v2 * v2;
    #pragma unroll
    for (int off = 16; off > 0; off >>= 1) q += __shfl_xor(q, off);
    __shared__ float sq[8];
    const int w = tid >> 5, ln = tid & 31;
    if (ln == 0) sq[w] = q;
    __syncthreads();
    if (w == 0) {
        float b = (ln < 8) ? sq[ln] : 0.f;
        #pragma unroll
        for (int off = 4; off > 0; off >>= 1) b += __shfl_xor(b, off);
        if (ln == 0) csq[row] = b;
    }

    __bf16* oh = chi + (size_t)row * D_DIM;
    __bf16* ol = clo + (size_t)row * D_DIM;
    float vals[3] = { v0, v1, v2 };
    #pragma unroll
    for (int i = 0; i < 3; ++i) {
        const int d = tid + i * 256;
        float v  = vals[i];
        __bf16 h = (__bf16)v;
        oh[d] = h;
        ol[d] = (__bf16)(v - (float)h);
    }
}

// ------------- Kernel 3: split-bf16 WMMA GEMM + fused argmin epilogue -------------
// One wave owns 16 rows; A is loaded ONCE into explicit register arrays (~384 VGPRs,
// gfx1250 extended-VGPR file) and lives there across the whole N sweep.
// B strips (16 codes x 768, hi+lo) are double-buffered in LDS via async copies.
__global__ __launch_bounds__(256) void gemm_argmin_kernel(
    const __bf16* __restrict__ Ehi, const __bf16* __restrict__ Elo,
    const __bf16* __restrict__ Chi, const __bf16* __restrict__ Clo,
    const float* __restrict__ Csq, int* __restrict__ out)
{
    extern __shared__ char smem[];
    float* s_csq = (float*)smem;
    char*  sB    = smem + LDS_CSQ;

    for (int i = threadIdx.x; i < K_CODES; i += 256) s_csq[i] = Csq[i];

    const int wave = threadIdx.x >> 5;
    const int lane = threadIdx.x & 31;
    const int lh   = lane & 15;
    const int sel  = lane >> 4;

    const int    rbase = (blockIdx.x * 8 + wave) * 16;
    const size_t arow  = (size_t)(rbase + lh) * D_DIM;

    // ---- Load the full A tile (16 rows x 768, hi+lo) into registers, once ----
    v16bf a_hi[K_STEPS], a_lo[K_STEPS];
    #pragma unroll
    for (int kk = 0; kk < K_STEPS; ++kk) {
        const int k = kk * 32;
        union { v16bf v; v8bf h[2]; } th, tl;
        th.h[0] = *(const v8bf*)(Ehi + arow + k +      8 * sel);
        th.h[1] = *(const v8bf*)(Ehi + arow + k + 16 + 8 * sel);
        tl.h[0] = *(const v8bf*)(Elo + arow + k +      8 * sel);
        tl.h[1] = *(const v8bf*)(Elo + arow + k + 16 + 8 * sel);
        a_hi[kk] = th.v;
        a_lo[kk] = tl.v;
    }

    // Stage one 16-code strip (hi+lo) into LDS buffer `buf`.
    // Each wave copies 4 row-planes = 12 async B128 ops (512 B per op, 32 lanes x 16 B).
    auto stage_tile = [&](int nb, int buf) {
        #pragma unroll
        for (int q = 0; q < 4; ++q) {
            const int rp  = wave * 4 + q;   // 0..31 row-plane index
            const int pl  = rp >> 4;        // 0 = hi plane, 1 = lo plane
            const int row = rp & 15;
            const char* gsrc = (const char*)(pl ? Clo : Chi) +
                               (size_t)(nb + row) * ROW_BYTES;
            char* ldst = sB + buf * LDS_BUF + pl * LDS_PLANE + row * LDS_ROW_STRIDE;
            #pragma unroll
            for (int i = 0; i < 3; ++i)
                async_cp16(gsrc + i * 512 + lane * 16, ldst + i * 512 + lane * 16);
        }
    };

    float minv[8];
    int   mini[8];
    #pragma unroll
    for (int r = 0; r < 8; ++r) { minv[r] = 3.4e38f; mini[r] = 0; }

    stage_tile(0, 0);

    int buf = 0;
    for (int nb = 0; nb < K_CODES; nb += 16, buf ^= 1) {
        __syncthreads();                       // all waves done reading buf^1 (and s_csq ready)
        if (nb + 16 < K_CODES) {
            stage_tile(nb + 16, buf ^ 1);      // prefetch next strip into the idle buffer
            WAIT_ASYNC(12);                    // my 12 copies for the *current* strip are done
        } else {
            WAIT_ASYNC(0);
        }
        __syncthreads();                       // every wave's copies for this strip complete

        // B fragment base for this lane: column (code) lh, K-octet chosen by sel
        const char* bh_row = sB + buf * LDS_BUF + lh * LDS_ROW_STRIDE + 16 * sel;
        const char* bl_row = bh_row + LDS_PLANE;

        v8f acc = {};
        #pragma unroll
        for (int kk = 0; kk < K_STEPS; ++kk) {
            const int k = kk * 32;
            union { v16bf v; v8bf h[2]; } bh, bl;
            // B from LDS (two 16B ds loads per fragment; bank-conflict-free padding)
            bh.h[0] = *(const v8bf*)(bh_row + k * 2);
            bh.h[1] = *(const v8bf*)(bh_row + k * 2 + 32);
            bl.h[0] = *(const v8bf*)(bl_row + k * 2);
            bl.h[1] = *(const v8bf*)(bl_row + k * 2 + 32);
            // error-compensated bf16: hi*hi + hi*lo + lo*hi, f32 accumulate
            acc = __builtin_amdgcn_wmma_f32_16x16x32_bf16(false, a_lo[kk], false, bh.v,
                                                          (short)0, acc, false, false);
            acc = __builtin_amdgcn_wmma_f32_16x16x32_bf16(false, a_hi[kk], false, bl.v,
                                                          (short)0, acc, false, false);
            acc = __builtin_amdgcn_wmma_f32_16x16x32_bf16(false, a_hi[kk], false, bh.v,
                                                          (short)0, acc, false, false);
        }

        // score = ||C_k||^2 - 2*cross (argmin-equivalent to clamped sqrt distance)
        const int   col = nb + lh;
        const float cs  = s_csq[col];
        #pragma unroll
        for (int r = 0; r < 8; ++r) {
            const float sc = cs - 2.0f * acc[r];
            if (sc < minv[r]) { minv[r] = sc; mini[r] = col; }  // strict '<' keeps first index
        }
    }

    // butterfly argmin across the 16 lanes of each half (rows 0..7 / 8..15)
    #pragma unroll
    for (int r = 0; r < 8; ++r) {
        float v = minv[r];
        int   i = mini[r];
        #pragma unroll
        for (int off = 8; off > 0; off >>= 1) {
            const float ov = __shfl_xor(v, off);
            const int   oi = __shfl_xor(i, off);
            if (ov < v || (ov == v && oi < i)) { v = ov; i = oi; }
        }
        mini[r] = i;
    }
    if (lh == 0) {
        #pragma unroll
        for (int r = 0; r < 8; ++r) out[rbase + sel * 8 + r] = mini[r];
    }
}

// ---------------------------------- launcher ----------------------------------
extern "C" void kernel_launch(void* const* d_in, const int* in_sizes, int n_in,
                              void* d_out, int out_size, void* d_ws, size_t ws_size,
                              hipStream_t stream) {
    const float* E = (const float*)d_in[0];   // [32,1500,768] f32
    const float* C = (const float*)d_in[1];   // [2048,768]    f32
    int* out = (int*)d_out;                   // [32,1,1500]   int32 (flat 48000)

    char* ws = (char*)d_ws;
    // workspace layout (bytes), all 32B-aligned:
    //   e_hi : 73,728,000 | e_lo : 73,728,000 | c_hi : 3,145,728 | c_lo : 3,145,728 | c_sq : 8,192
    __bf16* ehi = (__bf16*)(ws);
    __bf16* elo = (__bf16*)(ws + 73728000ull);
    __bf16* chi = (__bf16*)(ws + 147456000ull);
    __bf16* clo = (__bf16*)(ws + 150601728ull);
    float*  csq = (float*) (ws + 153747456ull);

    ln_split_kernel   <<<N_ROWS,  256, 0, stream>>>(E, ehi, elo);
    prep_c_kernel     <<<K_CODES, 256, 0, stream>>>(C, chi, clo, csq);
    gemm_argmin_kernel<<<N_ROWS / 128, 256, SMEM_BYTES, stream>>>(ehi, elo, chi, clo, csq, out);
}
